// Decoder_78700980732140
// MI455X (gfx1250) — compile-verified, matrix operand order checked by reading, not run on hardware
//
#include <hip/hip_runtime.h>
#include <hip/hip_bf16.h>

// ---------------------------------------------------------------------------
// Types for CDNA5 WMMA (gfx1250, wave32)
// ---------------------------------------------------------------------------
typedef __attribute__((ext_vector_type(16))) __bf16    v16bf;
typedef __attribute__((ext_vector_type(8)))  float     v8f;
typedef __attribute__((ext_vector_type(4)))  unsigned  u32x4;

union Frag { v16bf v; u32x4 u[2]; };
union Pack8 { __bf16 h[8]; u32x4 u; };

static __device__ __forceinline__ v8f bfwmma(v16bf a, v16bf b, v8f c) {
    return __builtin_amdgcn_wmma_f32_16x16x32_bf16(false, a, false, b,
                                                   (short)0, c, false, false);
}

static __device__ __forceinline__ __bf16 f2b(float f) {
    union { float f; unsigned u; } x{f};
    unsigned r = x.u + 0x7fffu + ((x.u >> 16) & 1u);   // round-to-nearest-even
    unsigned short h = (unsigned short)(r >> 16);
    __bf16 out;
    __builtin_memcpy(&out, &h, 2);
    return out;
}

// Problem constants
#define VV 32000
#define EE 512
#define HH 1024
#define TT 64
#define BB 64
#define KC 1536          // E + H (concat GEMM K)
#define G4 4096          // 4*H
#define NROW 4032        // 63 * 64 rows of hs / logits
#define NRT  252         // NROW / 16 row tiles
#define LSTM_BLOCKS 32
#define NB   500         // projection column blocks (64 cols each)

// ---------------------------------------------------------------------------
// Grid-wide barrier for the persistent LSTM kernel (32 co-resident blocks)
// ---------------------------------------------------------------------------
static __device__ __forceinline__ void gridbar(int* bar, int tgt) {
    __syncthreads();
    __threadfence();
    if (threadIdx.x == 0) {
        atomicAdd(bar, 1);
        volatile int* vb = bar;
        while (*vb < tgt) __builtin_amdgcn_s_sleep(2);
    }
    __syncthreads();
    __threadfence();
}

// Online softmax / argmax / label-logit update: ~1 exp per value.
static __device__ __forceinline__ void upd(float& m, float& s, int& a, float& l,
                                           float v, int c, int lab) {
    if (c == lab) l = v;
    if (v > m) { s = s * __expf(m - v) + 1.f; m = v; a = c; }
    else       { s += __expf(v - m); }
}

// ---------------------------------------------------------------------------
// 1) Embedding gather -> bf16 x-part of concat activation buffer A[t][b][0:512]
// ---------------------------------------------------------------------------
__global__ __launch_bounds__(256) void k_embed(const int* __restrict__ ids,
                                               const float* __restrict__ emb,
                                               __bf16* __restrict__ Abuf) {
    int idx = blockIdx.x * 256 + threadIdx.x;          // 63*64*512 total
    if (idx >= 63 * BB * EE) return;
    int t = idx / (BB * EE);
    int r = idx % (BB * EE);
    int b = r / EE, e = r % EE;
    int id = ids[t * BB + b];
    Abuf[((size_t)(t * BB + b)) * KC + e] = f2b(emb[(size_t)id * EE + e]);
}

// ---------------------------------------------------------------------------
// 2) Init h0 (bf16 into A[0] h-section) and reset the grid barrier counter
// ---------------------------------------------------------------------------
__global__ __launch_bounds__(256) void k_init(const float* __restrict__ h0,
                                              __bf16* __restrict__ Abuf,
                                              int* __restrict__ bar) {
    int idx = blockIdx.x * 256 + threadIdx.x;          // 64*1024
    if (idx == 0) bar[0] = 0;
    int b = idx >> 10, k = idx & 1023;
    Abuf[(size_t)b * KC + EE + k] = f2b(h0[idx]);
}

// ---------------------------------------------------------------------------
// 3) Weight conversions to bf16: Wcat = [W_ih | W_hh] (4096 x 1536), W_out
// ---------------------------------------------------------------------------
__global__ __launch_bounds__(256) void k_wcat(const float* __restrict__ Wih,
                                              const float* __restrict__ Whh,
                                              __bf16* __restrict__ Wcat) {
    int idx = blockIdx.x * 256 + threadIdx.x;          // 4096*1536
    int n = idx / KC, k = idx % KC;
    float v = (k < EE) ? Wih[(size_t)n * EE + k] : Whh[(size_t)n * HH + (k - EE)];
    Wcat[idx] = f2b(v);
}

__global__ __launch_bounds__(256) void k_wout(const float* __restrict__ W,
                                              __bf16* __restrict__ Wb) {
    int idx = blockIdx.x * 256 + threadIdx.x;          // 32000*1024
    Wb[idx] = f2b(W[idx]);
}

// ---------------------------------------------------------------------------
// 4) Persistent fused LSTM: all 63 steps in ONE launch.
//    Phase A: gates(64x4096) = A_t(64x1536) * Wcat^T  (WMMA bf16, f32 accum)
//    Phase B: elementwise cell update; c-state + biases live in registers.
//    Grid barrier (atomic + s_sleep spin) between phases.
// ---------------------------------------------------------------------------
__global__ __launch_bounds__(256) void k_lstm(__bf16* Abuf,
                                              const __bf16* __restrict__ Wcat,
                                              float* __restrict__ gates,
                                              const float* __restrict__ bih,
                                              const float* __restrict__ bhh,
                                              const float* __restrict__ c0,
                                              __bf16* __restrict__ hsB,
                                              int* bar) {
    const int tid = threadIdx.x, blk = blockIdx.x;
    const int lane = tid & 31, lr = lane & 15, half = lane >> 4, wv = tid >> 5;
    const int gw = blk * 8 + wv;            // global wave id [0,256)
    const int mtile = gw >> 6;              // [0,4)
    const int ng = (gw & 63) * 4;           // base N-tile of this wave's strip

    const int e0 = (blk * 256 + tid) * 8;   // 8 consecutive cell elements
    const int cb = e0 >> 10, ck = e0 & 1023;

    float c[8], bsi[8], bsf[8], bsg[8], bso[8];
#pragma unroll
    for (int j = 0; j < 8; ++j) {
        c[j] = c0[e0 + j];
        int k = ck + j;
        bsi[j] = bih[k]          + bhh[k];
        bsf[j] = bih[HH + k]     + bhh[HH + k];
        bsg[j] = bih[2*HH + k]   + bhh[2*HH + k];
        bso[j] = bih[3*HH + k]   + bhh[3*HH + k];
    }

    int tgt = 0;
    for (int t = 0; t < 63; ++t) {
        // ---- Phase A: gate GEMM ----
        const __bf16* arow = Abuf + ((size_t)t * BB + mtile * 16 + lr) * KC;
        const __bf16* br0 = Wcat + (size_t)((ng + 0) * 16 + lr) * KC;
        const __bf16* br1 = Wcat + (size_t)((ng + 1) * 16 + lr) * KC;
        const __bf16* br2 = Wcat + (size_t)((ng + 2) * 16 + lr) * KC;
        const __bf16* br3 = Wcat + (size_t)((ng + 3) * 16 + lr) * KC;
        Frag a, b0, b1, b2, b3;
        v8f ac0 = {}, ac1 = {}, ac2 = {}, ac3 = {};
        for (int kk = 0; kk < KC; kk += 32) {
            a.u[0]  = *(const u32x4*)(arow + kk + half * 8);
            a.u[1]  = *(const u32x4*)(arow + kk + 16 + half * 8);
            b0.u[0] = *(const u32x4*)(br0 + kk + half * 8);
            b0.u[1] = *(const u32x4*)(br0 + kk + 16 + half * 8);
            b1.u[0] = *(const u32x4*)(br1 + kk + half * 8);
            b1.u[1] = *(const u32x4*)(br1 + kk + 16 + half * 8);
            b2.u[0] = *(const u32x4*)(br2 + kk + half * 8);
            b2.u[1] = *(const u32x4*)(br2 + kk + 16 + half * 8);
            b3.u[0] = *(const u32x4*)(br3 + kk + half * 8);
            b3.u[1] = *(const u32x4*)(br3 + kk + 16 + half * 8);
            ac0 = bfwmma(a.v, b0.v, ac0);
            ac1 = bfwmma(a.v, b1.v, ac1);
            ac2 = bfwmma(a.v, b2.v, ac2);
            ac3 = bfwmma(a.v, b3.v, ac3);
        }
#pragma unroll
        for (int j = 0; j < 8; ++j) {
            int M = mtile * 16 + half * 8 + j;
            float* gr = gates + (size_t)M * G4 + lr;
            gr[(ng + 0) * 16] = ac0[j];
            gr[(ng + 1) * 16] = ac1[j];
            gr[(ng + 2) * 16] = ac2[j];
            gr[(ng + 3) * 16] = ac3[j];
        }
        tgt += LSTM_BLOCKS; gridbar(bar, tgt);

        // ---- Phase B: cell update ----
        {
            const float* g = gates + (size_t)cb * G4;
            Pack8 hv;
#pragma unroll
            for (int j = 0; j < 8; ++j) {
                int k = ck + j;
                float gi = g[k]          + bsi[j];
                float gf = g[HH + k]     + bsf[j];
                float gg = g[2*HH + k]   + bsg[j];
                float go = g[3*HH + k]   + bso[j];
                float i_ = 1.f / (1.f + __expf(-gi));
                float f_ = 1.f / (1.f + __expf(-gf));
                float o_ = 1.f / (1.f + __expf(-go));
                c[j] = f_ * c[j] + i_ * tanhf(gg);
                hv.h[j] = f2b(o_ * tanhf(c[j]));
            }
            *(u32x4*)(hsB + ((size_t)t * BB + cb) * HH + ck) = hv.u;
            if (t < 62)
                *(u32x4*)(Abuf + ((size_t)(t + 1) * BB + cb) * KC + EE + ck) = hv.u;
        }
        tgt += LSTM_BLOCKS; gridbar(bar, tgt);
    }
}

// ---------------------------------------------------------------------------
// 5) Projection, column-partitioned: 500 blocks each own 64 vocab columns.
//    The block's W_out slice (64x1024 bf16, padded stride) is parked in LDS
//    ONCE (-> W_out read exactly once from HBM/L2 in total); the small hs
//    matrix (8 MB, L2-resident) is streamed. Each wave owns whole 16-row
//    tiles: 4 N-tile accumulators (128 WMMA/tile), online softmax state per
//    row, shfl_xor reduce across 16 lanes, flush per-(row, block) partial.
// ---------------------------------------------------------------------------
__global__ __launch_bounds__(256) void k_proj(const __bf16* __restrict__ hsB,
                                              const __bf16* __restrict__ WoutB,
                                              const float* __restrict__ bout,
                                              const int* __restrict__ ids,
                                              float* __restrict__ Pmax,
                                              float* __restrict__ Psum,
                                              int*   __restrict__ Parg,
                                              float* __restrict__ Plab) {
    __shared__ __bf16 Bs[64 * 1032];                   // 129 KB, conflict-free pad

    const int tid = threadIdx.x;
    const int blk = blockIdx.x;                        // column block [0,500)
    const int nbase = blk * 64;                        // first vocab column

    // park this block's W_out slice in LDS (once)
    for (int i = tid; i < 64 * 64; i += 256) {
        int row = i >> 6, ch = i & 63;
        *((u32x4*)(Bs + row * 1032) + ch) =
            *((const u32x4*)(WoutB + (size_t)(nbase + row) * HH) + ch);
    }
    __syncthreads();

    const int wv = tid >> 5, lane = tid & 31, lr = lane & 15, half = lane >> 4;

    const __bf16* b0r = Bs + (0 * 16 + lr) * 1032;
    const __bf16* b1r = Bs + (1 * 16 + lr) * 1032;
    const __bf16* b2r = Bs + (2 * 16 + lr) * 1032;
    const __bf16* b3r = Bs + (3 * 16 + lr) * 1032;
    const float bo0 = bout[nbase + 0 * 16 + lr];
    const float bo1 = bout[nbase + 1 * 16 + lr];
    const float bo2 = bout[nbase + 2 * 16 + lr];
    const float bo3 = bout[nbase + 3 * 16 + lr];
    const int c0 = nbase + 0 * 16 + lr, c1 = nbase + 1 * 16 + lr;
    const int c2 = nbase + 2 * 16 + lr, c3 = nbase + 3 * 16 + lr;

    for (int rt = wv; rt < NRT; rt += 8) {             // whole 16-row tiles per wave
        const __bf16* ar = hsB + (size_t)(rt * 16 + lr) * HH;
        if (rt + 8 < NRT)                              // -> global_prefetch_b8
            __builtin_prefetch(hsB + (size_t)((rt + 8) * 16 + lr) * HH, 0, 1);

        Frag a, b0, b1, b2, b3;
        v8f acc0 = {}, acc1 = {}, acc2 = {}, acc3 = {};
        for (int kk = 0; kk < HH; kk += 32) {
            a.u[0]  = *(const u32x4*)(ar  + kk + half * 8);
            a.u[1]  = *(const u32x4*)(ar  + kk + 16 + half * 8);
            b0.u[0] = *(const u32x4*)(b0r + kk + half * 8);
            b0.u[1] = *(const u32x4*)(b0r + kk + 16 + half * 8);
            b1.u[0] = *(const u32x4*)(b1r + kk + half * 8);
            b1.u[1] = *(const u32x4*)(b1r + kk + 16 + half * 8);
            b2.u[0] = *(const u32x4*)(b2r + kk + half * 8);
            b2.u[1] = *(const u32x4*)(b2r + kk + 16 + half * 8);
            b3.u[0] = *(const u32x4*)(b3r + kk + half * 8);
            b3.u[1] = *(const u32x4*)(b3r + kk + 16 + half * 8);
            acc0 = bfwmma(a.v, b0.v, acc0);
            acc1 = bfwmma(a.v, b1.v, acc1);
            acc2 = bfwmma(a.v, b2.v, acc2);
            acc3 = bfwmma(a.v, b3.v, acc3);
        }

        // per-row online softmax over this block's 64 columns
        float rmax[8], rsum[8], rlab[8]; int rarg[8], labv[8];
#pragma unroll
        for (int j = 0; j < 8; ++j) {
            rmax[j] = -1e30f; rsum[j] = 0.f; rlab[j] = -1e30f; rarg[j] = 0;
            labv[j] = ids[rt * 16 + half * 8 + j + BB];
        }
#pragma unroll
        for (int j = 0; j < 8; ++j) {
            upd(rmax[j], rsum[j], rarg[j], rlab[j], acc0[j] + bo0, c0, labv[j]);
            upd(rmax[j], rsum[j], rarg[j], rlab[j], acc1[j] + bo1, c1, labv[j]);
            upd(rmax[j], rsum[j], rarg[j], rlab[j], acc2[j] + bo2, c2, labv[j]);
            upd(rmax[j], rsum[j], rarg[j], rlab[j], acc3[j] + bo3, c3, labv[j]);
        }
        // reduce across the 16 columns held by the 16 lanes of each half
        for (int mask = 1; mask < 16; mask <<= 1) {
#pragma unroll
            for (int j = 0; j < 8; ++j) {
                float m2 = __shfl_xor(rmax[j], mask);
                float s2 = __shfl_xor(rsum[j], mask);
                int   a2 = __shfl_xor(rarg[j], mask);
                float l2 = __shfl_xor(rlab[j], mask);
                if (m2 > rmax[j] || (m2 == rmax[j] && a2 < rarg[j])) rarg[j] = a2;
                float nm = fmaxf(rmax[j], m2);
                rsum[j] = rsum[j] * __expf(rmax[j] - nm) + s2 * __expf(m2 - nm);
                rmax[j] = nm;
                rlab[j] = fmaxf(rlab[j], l2);
            }
        }
        if (lr == 0) {                                 // lanes 0 & 16: 8 rows each
#pragma unroll
            for (int j = 0; j < 8; ++j) {
                size_t pi = (size_t)(rt * 16 + half * 8 + j) * NB + blk;
                Pmax[pi] = rmax[j]; Psum[pi] = rsum[j];
                Plab[pi] = rlab[j]; Parg[pi] = rarg[j];
            }
        }
    }
}

// ---------------------------------------------------------------------------
// 6) Combine the 500 per-column-block partials for every row.
// ---------------------------------------------------------------------------
__global__ __launch_bounds__(256) void k_comb(const float* __restrict__ Pmax,
                                              const float* __restrict__ Psum,
                                              const int*   __restrict__ Parg,
                                              const float* __restrict__ Plab,
                                              int* __restrict__ pred,
                                              float* __restrict__ lse,
                                              float* __restrict__ labl) {
    int r = blockIdx.x * 256 + threadIdx.x;
    if (r >= NROW) return;
    float M = -1e30f, S = 0.f, L = -1e30f; int A = 0;
    for (int w = 0; w < NB; ++w) {
        size_t pi = (size_t)r * NB + w;
        float m2 = Pmax[pi], s2 = Psum[pi], l2 = Plab[pi];
        int a2 = Parg[pi];
        if (m2 > M || (m2 == M && a2 < A)) A = a2;
        float nm = fmaxf(M, m2);
        S = S * __expf(M - nm) + s2 * __expf(m2 - nm);
        M = nm;
        L = fmaxf(L, l2);
    }
    pred[r] = A;
    lse[r]  = M + __logf(S);
    labl[r] = L;
}

// ---------------------------------------------------------------------------
// 7) Finalize: masked NLL loss (1 elem) + result matrix (64x64) into d_out
// ---------------------------------------------------------------------------
__global__ __launch_bounds__(256) void k_final(const int* __restrict__ pred,
                                               const float* __restrict__ lse,
                                               const float* __restrict__ labl,
                                               const int* __restrict__ ids,
                                               float* __restrict__ out) {
    __shared__ float ps[63];
    int tid = threadIdx.x;
    if (tid < 63) {
        float num = 0.f, den = 0.f;
        for (int b = 0; b < BB; ++b) {
            int r = tid * BB + b;
            int lab = ids[r + BB];
            float mask = (lab != 0) ? 1.f : 0.f;
            num += (lse[r] - labl[r]) * mask;
            den += mask;
        }
        ps[tid] = num / fmaxf(den, 1.f);
    }
    __syncthreads();
    if (tid == 0) {
        float s = 0.f;
        for (int t = 0; t < 63; ++t) s += ps[t];
        out[0] = s;
    }
    for (int i = tid; i < TT * BB; i += 256) {
        int row = i >> 6, b = i & 63;
        out[1 + i] = (row == 0) ? 1.0f : (float)pred[(row - 1) * BB + b];
    }
}

// ---------------------------------------------------------------------------
// Host launcher
// ---------------------------------------------------------------------------
extern "C" void kernel_launch(void* const* d_in, const int* in_sizes, int n_in,
                              void* d_out, int out_size, void* d_ws, size_t ws_size,
                              hipStream_t stream) {
    const int*   ids  = (const int*)d_in[0];
    const float* h0   = (const float*)d_in[3];
    const float* c0   = (const float*)d_in[4];
    const float* emb  = (const float*)d_in[5];
    const float* Wih  = (const float*)d_in[6];
    const float* Whh  = (const float*)d_in[7];
    const float* bih  = (const float*)d_in[8];
    const float* bhh  = (const float*)d_in[9];
    const float* Wout = (const float*)d_in[10];
    const float* bout = (const float*)d_in[11];
    float* out = (float*)d_out;

    char* ws = (char*)d_ws;
    size_t off = 0;
    auto alloc = [&](size_t bytes) -> void* {
        void* p = ws + off;
        off = (off + bytes + 255) & ~(size_t)255;
        return p;
    };
    __bf16* Abuf  = (__bf16*)alloc((size_t)63 * BB * KC * 2);     // concat [x|h] acts
    __bf16* Wcat  = (__bf16*)alloc((size_t)G4 * KC * 2);          // [W_ih|W_hh] bf16
    __bf16* WoutB = (__bf16*)alloc((size_t)VV * HH * 2);          // W_out bf16
    __bf16* hsB   = (__bf16*)alloc((size_t)NROW * HH * 2);        // hidden states bf16
    float*  gates = (float*)alloc((size_t)BB * G4 * 4);
    int*    pred  = (int*)alloc((size_t)NROW * 4);
    float*  lseb  = (float*)alloc((size_t)NROW * 4);
    float*  labl  = (float*)alloc((size_t)NROW * 4);
    int*    bar   = (int*)alloc(256);
    float*  Pmax  = (float*)alloc((size_t)NROW * NB * 4);         // projection partials
    float*  Psum  = (float*)alloc((size_t)NROW * NB * 4);
    int*    Parg  = (int*)alloc((size_t)NROW * NB * 4);
    float*  Plab  = (float*)alloc((size_t)NROW * NB * 4);

    k_embed<<<(63 * BB * EE + 255) / 256, 256, 0, stream>>>(ids, emb, Abuf);
    k_init <<<(BB * HH) / 256, 256, 0, stream>>>(h0, Abuf, bar);
    k_wcat <<<(G4 * KC) / 256, 256, 0, stream>>>(Wih, Whh, Wcat);
    k_wout <<<(VV * HH) / 256, 256, 0, stream>>>(Wout, WoutB);

    k_lstm <<<LSTM_BLOCKS, 256, 0, stream>>>(Abuf, Wcat, gates, bih, bhh, c0, hsB, bar);

    k_proj <<<NB, 256, 0, stream>>>(hsB, WoutB, bout, ids, Pmax, Psum, Parg, Plab);
    k_comb <<<(NROW + 255) / 256, 256, 0, stream>>>(Pmax, Psum, Parg, Plab, pred, lseb, labl);
    k_final<<<1, 256, 0, stream>>>(pred, lseb, labl, ids, out);
}
